// MultiHeadAttention_8151847928422
// MI455X (gfx1250) — matrix operand order repaired and checked by reading, not compiled, not run on hardware
//
#include <hip/hip_runtime.h>
#include <hip/hip_bf16.h>
#include <cstdint>

typedef _Float16 v16h __attribute__((ext_vector_type(16)));
typedef _Float16 v8h  __attribute__((ext_vector_type(8)));
typedef float    v8f  __attribute__((ext_vector_type(8)));
typedef float    v4f  __attribute__((ext_vector_type(4)));
typedef unsigned int u32x4 __attribute__((ext_vector_type(4)));
typedef int      i32x8 __attribute__((ext_vector_type(8)));
typedef int      i32x4 __attribute__((ext_vector_type(4)));

#define B_   4
#define S_   2048
#define DM   1024
#define NH   16
#define HD   64
#define MROWS (B_*S_)   // 8192

static __device__ __forceinline__ v8f wmma_f16(v16h a, v16h b, v8f c) {
  // 8 args: (neg_a, A, neg_b, B, c_mod, C, reuse_a, reuse_b)
  return __builtin_amdgcn_wmma_f32_16x16x32_f16(false, a, false, b, (short)0, c, false, false);
}

// -------------------------------------------------------------- TDM helper
// 2D tensor tile -> LDS via Tensor Data Mover (D# per CDNA5 ISA ch.8).
// pad_interval: 0=2,1=4,2=8,3=16,4=32.. dwords; pad_amount: code+1 dwords.
static __device__ __forceinline__ void tdm_load_2d(
    unsigned long long gaddr, unsigned int lds_off,
    unsigned int td0, unsigned int td1,          // tensor dims (elements)
    unsigned int tile0, unsigned int tile1,      // tile dims (elements)
    unsigned int stride0,                        // dim0 stride (elements)
    unsigned int dsz_log,                        // 0=1B,1=2B,2=4B
    unsigned int pad_interval, unsigned int pad_amount) {
  u32x4 g0;
  g0[0] = 1u;                                          // count=1, user D#
  g0[1] = lds_off;                                     // lds_addr
  g0[2] = (unsigned int)(gaddr & 0xFFFFFFFFu);         // global_addr[31:0]
  g0[3] = (unsigned int)((gaddr >> 32) & 0x1FFFFFFu)   // global_addr[56:32]
        | (2u << 30);                                  // type=2 (image)
  i32x8 g1;
  g1[0] = (int)((dsz_log << 16) | (1u << 20)           // data_size | pad_enable
        | (pad_interval << 22) | (pad_amount << 25));
  g1[1] = (int)((td0 & 0xFFFFu) << 16);                // tensor_dim0[15:0]
  g1[2] = (int)((td0 >> 16) | ((td1 & 0xFFFFu) << 16));// dim0.hi | dim1.lo
  g1[3] = (int)((td1 >> 16) | (tile0 << 16));          // dim1.hi | tile_dim0
  g1[4] = (int)(tile1 & 0xFFFFu);                      // tile_dim1 (tile_dim2=0)
  g1[5] = (int)stride0;                                // dim0_stride low32
  g1[6] = 0;
  g1[7] = 0;
  i32x4 z4 = {0, 0, 0, 0};                             // groups 2/3: 2D tensor
  i32x8 z8 = {0, 0, 0, 0, 0, 0, 0, 0};
  __builtin_amdgcn_tensor_load_to_lds(g0, g1, z4, z4, z8, 0);
}

// ---------------------------------------------------------------- f32 -> f16
__global__ __launch_bounds__(256) void f32_to_f16_kernel(const float* __restrict__ s,
                                                         _Float16* __restrict__ d, int n) {
  int i = blockIdx.x * blockDim.x + threadIdx.x;
  if (i < n) d[i] = (_Float16)s[i];
}

// ------------------------------------------------------- GEMM: C = A * B^T
// A: (8192,1024) f32 row-major; Bh: (1024,1024) f16 row-major (the weight W,
// since y = x @ W^T). One wave -> 16x64 tile; all 8 waves of a block share m0.
// TDM stages per k-chunk: the shared 16x32 f32 A panel (wave 0) and each
// wave's 64x32-half B panel, double-buffered, padded LDS rows.
// mode 0: f16 row-major out. mode 1: scatter into Vt[b,h,d,s] f16.
// mode 2: f32 row-major out (d_out).
__global__ __launch_bounds__(256) void gemm_xwT_kernel(const float* __restrict__ A,
                                                       const _Float16* __restrict__ Bh,
                                                       void* __restrict__ Cout, int mode) {
  const int Kd = DM, Nd = DM;
  __shared__ __align__(16) float    Abuf[2][16 * 36];        // 36-dword row stride
  __shared__ __align__(16) _Float16 Bbuf[8][2][64 * 40];     // 40-half row stride
  int wid  = threadIdx.x >> 5;
  int lane = threadIdx.x & 31;
  int nn = lane & 15, kh = lane >> 4;
  int tile = blockIdx.x * 8 + wid;
  int tn = Nd / 64;                 // 16 tiles along N
  int m0 = (tile / tn) * 16;
  int n0 = (tile % tn) * 64;

  const unsigned long long Aga = (unsigned long long)(uintptr_t)A + (unsigned long long)m0 * Kd * 4ull;
  const unsigned long long Bga = (unsigned long long)(uintptr_t)Bh + (unsigned long long)n0 * Kd * 2ull;

  // issue chunk 0 DMAs
  if (wid == 0)
    tdm_load_2d(Aga, (unsigned int)(uintptr_t)&Abuf[0][0], Kd, MROWS, 32, 16, Kd, 2u, 4u, 3u);
  tdm_load_2d(Bga, (unsigned int)(uintptr_t)&Bbuf[wid][0][0], Kd, Nd, 32, 64, Kd, 1u, 3u, 3u);

  v8f acc[4] = {};
  const int NCH = Kd / 32;
  for (int i = 0; i < NCH; ++i) {
    __builtin_amdgcn_s_wait_tensorcnt(0);   // chunk i resident in LDS
    __syncthreads();                        // publish A panel; prior buf reads done
    int cur = i & 1, nxt = cur ^ 1;
    if (i + 1 < NCH) {                      // overlap chunk i+1 DMA with compute
      unsigned long long ko = (unsigned long long)(i + 1) * 32ull;
      if (wid == 0)
        tdm_load_2d(Aga + ko * 4ull, (unsigned int)(uintptr_t)&Abuf[nxt][0],
                    Kd, MROWS, 32, 16, Kd, 2u, 4u, 3u);
      tdm_load_2d(Bga + ko * 2ull, (unsigned int)(uintptr_t)&Bbuf[wid][nxt][0],
                  Kd, Nd, 32, 64, Kd, 1u, 3u, 3u);
    }
    // A fragment: lane (m=nn, kh) holds K = kh*16+0..15
    const float* ar = &Abuf[cur][nn * 36 + kh * 16];
    v4f a0 = *(const v4f*)ar;
    v4f a1 = *(const v4f*)(ar + 4);
    v4f a2 = *(const v4f*)(ar + 8);
    v4f a3 = *(const v4f*)(ar + 12);
    v16h af;
#pragma unroll
    for (int j = 0; j < 4; ++j) {
      af[j]      = (_Float16)a0[j];
      af[4 + j]  = (_Float16)a1[j];
      af[8 + j]  = (_Float16)a2[j];
      af[12 + j] = (_Float16)a3[j];
    }
#pragma unroll
    for (int t = 0; t < 4; ++t) {
      // B fragment: lane (n=nn, kh) holds B[kh*16+j][n] = W[n0+t*16+nn][k+kh*16+j]
      v16h bf = *(const v16h*)&Bbuf[wid][cur][(t * 16 + nn) * 40 + kh * 16];
      acc[t] = wmma_f16(af, bf, acc[t]);
    }
  }

  if (mode == 0) {
    _Float16* C = (_Float16*)Cout;
#pragma unroll
    for (int t = 0; t < 4; ++t)
#pragma unroll
      for (int i = 0; i < 8; ++i)
        C[(size_t)(m0 + i + 8 * kh) * Nd + n0 + t * 16 + nn] = (_Float16)acc[t][i];
  } else if (mode == 1) {
    _Float16* Vt = (_Float16*)Cout;
#pragma unroll
    for (int t = 0; t < 4; ++t)
#pragma unroll
      for (int i = 0; i < 8; ++i) {
        int m = m0 + i + 8 * kh;
        int b = m >> 11;            // S_ = 2048
        int s = m & (S_ - 1);
        int e = n0 + t * 16 + nn;
        int h = e >> 6;             // HD = 64
        int d = e & 63;
        Vt[(((size_t)b * NH + h) * HD + d) * S_ + s] = (_Float16)acc[t][i];
      }
  } else {
    float* C = (float*)Cout;
#pragma unroll
    for (int t = 0; t < 4; ++t)
#pragma unroll
      for (int i = 0; i < 8; ++i)
        C[(size_t)(m0 + i + 8 * kh) * Nd + n0 + t * 16 + nn] = acc[t][i];
  }
}

// ------------------------------------------------------------------- RoPE
__global__ __launch_bounds__(256) void rope_kernel(const _Float16* __restrict__ Qraw,
                                                   const _Float16* __restrict__ Kraw,
                                                   const int* __restrict__ pos,
                                                   _Float16* __restrict__ Qr,
                                                   _Float16* __restrict__ Kr) {
  int idx = blockIdx.x * blockDim.x + threadIdx.x;  // B*NH*S*32 = 4M
  int i = idx & 31;                 // pair index within head dim
  int s = (idx >> 5) & (S_ - 1);
  int h = (idx >> 16) & (NH - 1);
  int b = idx >> 20;
  float p = (float)pos[b * S_ + s];
  float inv = __expf(-((float)(2 * i) / (float)HD) * 9.210340371976184f); // ln(10000)
  float ang = p * inv;
  float cs = __cosf(ang), sn = __sinf(ang);
  size_t rbase = ((size_t)b * S_ + s) * DM + h * HD + 2 * i;
  size_t wbase = (((size_t)(b * NH + h) * S_) + s) * HD + 2 * i;
  {
    float e = (float)Qraw[rbase], o = (float)Qraw[rbase + 1];
    Qr[wbase]     = (_Float16)(e * cs - o * sn);
    Qr[wbase + 1] = (_Float16)(e * sn + o * cs);
  }
  {
    float e = (float)Kraw[rbase], o = (float)Kraw[rbase + 1];
    Kr[wbase]     = (_Float16)(e * cs - o * sn);
    Kr[wbase + 1] = (_Float16)(e * sn + o * cs);
  }
}

// ------------------------------------------------- causal flash attention
__global__ __launch_bounds__(256) void attn_kernel(const _Float16* __restrict__ Q,
                                                   const _Float16* __restrict__ Kt,
                                                   const _Float16* __restrict__ Vt,
                                                   float* __restrict__ Ao) {
  __shared__ __align__(16) _Float16 plds[8][16 * 40];
  int wid  = threadIdx.x >> 5;
  int lane = threadIdx.x & 31;
  int nn = lane & 15, kh = lane >> 4;
  int tile = blockIdx.x * 8 + wid;
  int qt = tile & (S_ / 16 - 1);
  int h  = (tile >> 7) & (NH - 1);
  int b  = tile >> 11;
  int q0 = qt * 16;

  const _Float16* Qb = Q  + ((size_t)(b * NH + h) * S_ + q0) * HD;
  const _Float16* Kb = Kt + ((size_t)(b * NH + h) * S_) * HD;
  const _Float16* Vb = Vt + ((size_t)(b * NH + h) * HD) * S_;
  _Float16* pw = &plds[wid][0];

  v16h aq0, aq1;
  {
    const _Float16* qp = Qb + (size_t)nn * HD + kh * 16;
    v8h l0 = *(const v8h*)qp;
    v8h h0 = *(const v8h*)(qp + 8);
    v8h l1 = *(const v8h*)(qp + 32);
    v8h h1 = *(const v8h*)(qp + 40);
#pragma unroll
    for (int j = 0; j < 8; ++j) {
      aq0[j] = l0[j]; aq0[8 + j] = h0[j];
      aq1[j] = l1[j]; aq1[8 + j] = h1[j];
    }
  }

  v8f acc[4] = {};
  float mrow[8], lrow[8];
#pragma unroll
  for (int i = 0; i < 8; ++i) { mrow[i] = -1e30f; lrow[i] = 0.f; }

  for (int kt = 0; kt < q0 + 16; kt += 32) {
    v8f s1 = {}, s2 = {};
    const _Float16* kp1 = Kb + (size_t)(kt + nn) * HD;
    const _Float16* kp2 = Kb + (size_t)(kt + 16 + nn) * HD;
    s1 = wmma_f16(aq0, *(const v16h*)(kp1 + kh * 16), s1);
    s1 = wmma_f16(aq1, *(const v16h*)(kp1 + 32 + kh * 16), s1);
    s2 = wmma_f16(aq0, *(const v16h*)(kp2 + kh * 16), s2);
    s2 = wmma_f16(aq1, *(const v16h*)(kp2 + 32 + kh * 16), s2);

    float p1[8], p2[8], al[8];
#pragma unroll
    for (int i = 0; i < 8; ++i) {
      int qg = q0 + i + 8 * kh;
      float a = s1[i] * 0.125f;           // 1/sqrt(64)
      float c = s2[i] * 0.125f;
      if (kt + nn > qg)      a = -1e30f;
      if (kt + 16 + nn > qg) c = -1e30f;
      float mx = fmaxf(a, c);
#pragma unroll
      for (int off = 1; off < 16; off <<= 1) mx = fmaxf(mx, __shfl_xor(mx, off, 16));
      float mn = fmaxf(mrow[i], mx);
      float pa = __expf(a - mn), pb = __expf(c - mn);
      float sm = pa + pb;
#pragma unroll
      for (int off = 1; off < 16; off <<= 1) sm += __shfl_xor(sm, off, 16);
      al[i]   = __expf(mrow[i] - mn);
      lrow[i] = lrow[i] * al[i] + sm;
      mrow[i] = mn;
      p1[i] = pa; p2[i] = pb;
    }
#pragma unroll
    for (int t = 0; t < 4; ++t)
#pragma unroll
      for (int i = 0; i < 8; ++i) acc[t][i] *= al[i];

#pragma unroll
    for (int i = 0; i < 8; ++i) {
      pw[(i + 8 * kh) * 40 + nn]      = (_Float16)p1[i];
      pw[(i + 8 * kh) * 40 + 16 + nn] = (_Float16)p2[i];
    }
    v16h pa16;
    {
      v8h lo = *(const v8h*)(pw + nn * 40 + kh * 16);
      v8h hi = *(const v8h*)(pw + nn * 40 + kh * 16 + 8);
#pragma unroll
      for (int j = 0; j < 8; ++j) { pa16[j] = lo[j]; pa16[8 + j] = hi[j]; }
    }
#pragma unroll
    for (int t = 0; t < 4; ++t) {
      v16h bv = *(const v16h*)(Vb + (size_t)(t * 16 + nn) * S_ + kt + kh * 16);
      acc[t] = wmma_f16(pa16, bv, acc[t]);
    }
  }

  float rinv[8];
#pragma unroll
  for (int i = 0; i < 8; ++i) rinv[i] = 1.0f / lrow[i];
  float* Ab = Ao + ((size_t)b * S_ + q0) * DM + h * HD;
#pragma unroll
  for (int t = 0; t < 4; ++t)
#pragma unroll
    for (int i = 0; i < 8; ++i)
      Ab[(size_t)(i + 8 * kh) * DM + t * 16 + nn] = acc[t][i] * rinv[i];
}

// --------------------------------------------------------------- launcher
extern "C" void kernel_launch(void* const* d_in, const int* in_sizes, int n_in,
                              void* d_out, int out_size, void* d_ws, size_t ws_size,
                              hipStream_t stream) {
  (void)in_sizes; (void)n_in; (void)out_size; (void)ws_size;
  const float* x  = (const float*)d_in[0];
  const int*   tp = (const int*)d_in[1];
  const float* WQ = (const float*)d_in[2];
  const float* WK = (const float*)d_in[3];
  const float* WV = (const float*)d_in[4];
  const float* WO = (const float*)d_in[5];

  _Float16* WQh  = (_Float16*)d_ws;
  _Float16* WKh  = WQh + (size_t)DM * DM;
  _Float16* WVh  = WKh + (size_t)DM * DM;
  _Float16* WOh  = WVh + (size_t)DM * DM;
  _Float16* Qraw = WOh + (size_t)DM * DM;           // f16 [b,s,e]
  _Float16* Kraw = Qraw + (size_t)MROWS * DM;
  _Float16* Qr   = Kraw + (size_t)MROWS * DM;       // f16 [b,h,s,d]
  _Float16* Kr   = Qr   + (size_t)MROWS * DM;
  _Float16* Vt   = Kr   + (size_t)MROWS * DM;       // f16 [b,h,d,s]
  float*    attn = (float*)Qraw;                     // alias dead Qraw+Kraw (32MB f32)

  const int nW = DM * DM;
  f32_to_f16_kernel<<<(nW + 255) / 256, 256, 0, stream>>>(WQ, WQh, nW);
  f32_to_f16_kernel<<<(nW + 255) / 256, 256, 0, stream>>>(WK, WKh, nW);
  f32_to_f16_kernel<<<(nW + 255) / 256, 256, 0, stream>>>(WV, WVh, nW);
  f32_to_f16_kernel<<<(nW + 255) / 256, 256, 0, stream>>>(WO, WOh, nW);

  const int gemmBlocks = ((MROWS / 16) * (DM / 64)) / 8;   // 1024
  gemm_xwT_kernel<<<gemmBlocks, 256, 0, stream>>>(x, WQh, Qraw, 0);
  gemm_xwT_kernel<<<gemmBlocks, 256, 0, stream>>>(x, WKh, Kraw, 0);
  gemm_xwT_kernel<<<gemmBlocks, 256, 0, stream>>>(x, WVh, Vt, 1);

  rope_kernel<<<(B_ * NH * S_ * 32) / 256, 256, 0, stream>>>(Qraw, Kraw, tp, Qr, Kr);

  attn_kernel<<<(B_ * NH * (S_ / 16)) / 8, 256, 0, stream>>>(Qr, Kr, Vt, attn);

  gemm_xwT_kernel<<<gemmBlocks, 256, 0, stream>>>(attn, WOh, d_out, 2);
}
